// Llama4TextExperts_5394478924014
// MI455X (gfx1250) — matrix operand order repaired and checked by reading, not compiled, not run on hardware
//
#include <hip/hip_runtime.h>
#include <hip/hip_bf16.h>
#include <cstdint>

// ---------------------------------------------------------------------------
// MoE experts: out = (up * silu(gate)) @ down_w, bf16 WMMA path for MI455X.
//   E=8, TOK=1024 tokens/expert, H=2048, F=8192.
// Kernel 0: Xbf = bf16(X)                       (one-time convert)
// Kernel 1: h = up*silu(Xbf@gate, Xbf@up)       A via global_load_async_to_lds
// Kernel 2: out = h @ down_w                    A via TENSOR_LOAD_TO_LDS (TDM)
// Weight panels (f32, 1.6GB stream) are register-prefetched, converted with
// v_cvt_pk_bf16_f32 and staged transposed with packed b64 LDS stores.
// ---------------------------------------------------------------------------

#define NUM_EXPERTS 8
#define HID 2048
#define INTER 8192
#define TOK 1024

#define BM 128
#define BN 128
#define BK 32

typedef __attribute__((ext_vector_type(16))) __bf16 v16bf;
typedef __attribute__((ext_vector_type(8)))  __bf16 v8bf;
typedef __attribute__((ext_vector_type(4)))  __bf16 v4bf;
typedef __attribute__((ext_vector_type(8)))  float  v8f;
typedef __attribute__((ext_vector_type(4)))  unsigned int v4u;
typedef __attribute__((ext_vector_type(8)))  int    v8i;
typedef __attribute__((ext_vector_type(4)))  int    v4i;

__device__ __forceinline__ __bf16 f2bf(float f) { return (__bf16)f; }

// K-permutation so an A-matrix lane fragment (16 bf16) is one contiguous
// 32-byte LDS read (wave32 16-bit A layout):
//   K 0..7 ->0..7 | K16..23 ->8..15 | K8..15 ->16..23 | K24..31 ->24..31
__device__ __forceinline__ int permk(int k) {
    return ((k & 8) ? 16 : 0) + ((k & 16) ? 8 : 0) + (k & 7);
}

__device__ __forceinline__ v8f wmma_bf16(v16bf a, v16bf b, v8f c) {
    return __builtin_amdgcn_wmma_f32_16x16x32_bf16(
        false, a, false, b, (short)0, c, false, false);
}

__device__ __forceinline__ void wait_async0() {
    asm volatile("s_wait_asynccnt 0x0" ::: "memory");
}

__device__ __forceinline__ void async_copy_b128(unsigned lds_off, uint64_t gaddr) {
    asm volatile("global_load_async_to_lds_b128 %0, %1, off"
                 :: "v"(lds_off), "v"(gaddr) : "memory");
}

// TDM: load a [rows x 32] bf16 tile (row stride = `stride` elems) into LDS at
// lds_byte, packed row-major.  D# per CDNA5 ISA 8.3/8.4: count=1, type=2,
// data_size=1 (2B), tile_dim0=32, tile_dim1=rows, tensor_dim0_stride=stride.
__device__ __forceinline__ void tdm_load_tile(const __bf16* gsrc, unsigned lds_byte,
                                              int rows, int stride) {
    const uint64_t ga = (uint64_t)(uintptr_t)gsrc;
    v4u g0;
    g0.x = 1u;                                   // count=1, user descriptor
    g0.y = lds_byte;                             // lds_addr
    g0.z = (unsigned)ga;                         // global_addr[31:0]
    g0.w = (unsigned)(ga >> 32) | 0x80000000u;   // global_addr[56:32] | type=2
    v8i g1;
    g1[0] = 0x00010000;                          // data_size=1 (2 bytes)
    g1[1] = (int)((unsigned)INTER << 16);        // tensor_dim0[15:0] @ bits 63:48
    g1[2] = (int)(((unsigned)INTER >> 16) | ((unsigned)TOK << 16)); // dim0 hi | dim1 lo
    g1[3] = (int)((unsigned)BK << 16);           // tile_dim0=32 @ bits 127:112
    g1[4] = rows;                                // tile_dim1 (tile_dim2=0)
    g1[5] = stride;                              // tensor_dim0_stride[31:0]
    g1[6] = 0;                                   // dim0_stride hi | dim1_stride lo
    g1[7] = 0;
    v4i z4 = {0, 0, 0, 0};
    v8i z8 = {0, 0, 0, 0, 0, 0, 0, 0};
    __builtin_amdgcn_tensor_load_to_lds(g0, g1, z4, z4, z8, 0);
}

// ---------------------------------------------------------------------------
// Kernel 0: X (f32) -> Xbf (bf16), 8 elems/thread
// ---------------------------------------------------------------------------
__global__ __launch_bounds__(256)
void cvt_x_kernel(const float* __restrict__ X, __bf16* __restrict__ Xb) {
    const size_t i = ((size_t)blockIdx.x * 256 + threadIdx.x) * 8;
    float4 a = *(const float4*)(X + i);
    float4 b = *(const float4*)(X + i + 4);
    v8bf o = { f2bf(a.x), f2bf(a.y), f2bf(a.z), f2bf(a.w),
               f2bf(b.x), f2bf(b.y), f2bf(b.z), f2bf(b.w) };
    *(v8bf*)(Xb + i) = o;
}

// ---------------------------------------------------------------------------
// Kernel 1: gate/up GEMMs fused with SiLU.  grid = (F/BN, TOK/BM, E)
// ---------------------------------------------------------------------------
__global__ __launch_bounds__(256)
void moe_gateup_kernel(const __bf16* __restrict__ Xb,
                       const float* __restrict__ Gw,
                       const float* __restrict__ Uw,
                       __bf16* __restrict__ Hws) {
    __shared__ __align__(32) __bf16 sA [2][BM * BK];   // row-major [m][k]
    __shared__ __align__(32) __bf16 sBg[2][BN * BK];   // transposed [n][k]
    __shared__ __align__(32) __bf16 sBu[2][BN * BK];

    const int e  = blockIdx.z;
    const int m0 = blockIdx.y * BM;
    const int n0 = blockIdx.x * BN;

    const __bf16* Xe = Xb + (size_t)e * TOK * HID + (size_t)m0 * HID;
    const float*  Ge = Gw + (size_t)e * HID * INTER;
    const float*  Ue = Uw + (size_t)e * HID * INTER;

    const int t    = threadIdx.x;
    const int lane = t & 31;
    const int wave = t >> 5;
    const int wm   = wave & 3;        // 4 waves along M: 32 rows each
    const int wn   = wave >> 2;       // 2 waves along N: 64 cols each
    const int half = lane >> 4;
    const int l16  = lane & 15;

    const int bk4 = (t >> 5) << 2;    // 0,4,...,28
    const int bn4 = (t & 31) << 2;    // 0,4,...,124

    float gr[4][4], ur[4][4];         // B prefetch regs [krow][ncol]

    auto issue_asyncA = [&](int k0, int p) {   // 2 x 16B per thread, no VGPR data
#pragma unroll
        for (int r = 0; r < 2; ++r) {
            const int q = t + 256 * r, row = q >> 2, kq = (q & 3) << 3;
            async_copy_b128(
                (unsigned)(uintptr_t)(&sA[p][row * BK + kq]),
                (uint64_t)(uintptr_t)(Xe + (size_t)row * HID + k0 + kq));
        }
    };
    auto loadB = [&](int k0) {
#pragma unroll
        for (int r = 0; r < 4; ++r) {
            const size_t go = (size_t)(k0 + bk4 + r) * INTER + n0 + bn4;
            float4 g = *(const float4*)(Ge + go);
            float4 u = *(const float4*)(Ue + go);
            gr[r][0] = g.x; gr[r][1] = g.y; gr[r][2] = g.z; gr[r][3] = g.w;
            ur[r][0] = u.x; ur[r][1] = u.y; ur[r][2] = u.z; ur[r][3] = u.w;
        }
    };
    auto storeB = [&](int p) {
#pragma unroll
        for (int c = 0; c < 4; ++c) {
            v4bf g = { f2bf(gr[0][c]), f2bf(gr[1][c]), f2bf(gr[2][c]), f2bf(gr[3][c]) };
            v4bf u = { f2bf(ur[0][c]), f2bf(ur[1][c]), f2bf(ur[2][c]), f2bf(ur[3][c]) };
            *(v4bf*)(&sBg[p][(bn4 + c) * BK + bk4]) = g;  // 4 k's contiguous
            *(v4bf*)(&sBu[p][(bn4 + c) * BK + bk4]) = u;
        }
    };

    v8f accG[2][4], accU[2][4];
#pragma unroll
    for (int i = 0; i < 2; ++i)
#pragma unroll
        for (int j = 0; j < 4; ++j) { accG[i][j] = {}; accU[i][j] = {}; }

    issue_asyncA(0, 0);
    loadB(0);
    int p = 0;
    for (int k0 = 0; k0 < HID; k0 += BK) {
        storeB(p);
        wait_async0();                 // this wave's A-tile chunks for buf p landed
        __syncthreads();
        if (k0 + BK < HID) {
            issue_asyncA(k0 + BK, p ^ 1);    // overlaps with WMMA below
            loadB(k0 + BK);
        }

        v16bf a0 = *(const v16bf*)(&sA[p][(wm * 32 + l16) * BK + half * 16]);
        v16bf a1 = *(const v16bf*)(&sA[p][(wm * 32 + 16 + l16) * BK + half * 16]);
#pragma unroll
        for (int j = 0; j < 4; ++j) {
            const int nrow = (wn * 64 + j * 16 + l16) * BK + half * 16;
            v16bf bg = *(const v16bf*)(&sBg[p][nrow]);
            v16bf bu = *(const v16bf*)(&sBu[p][nrow]);
            accG[0][j] = wmma_bf16(a0, bg, accG[0][j]);
            accG[1][j] = wmma_bf16(a1, bg, accG[1][j]);
            accU[0][j] = wmma_bf16(a0, bu, accU[0][j]);
            accU[1][j] = wmma_bf16(a1, bu, accU[1][j]);
        }
        p ^= 1;
    }

    // epilogue: h = up * silu(gate), store bf16
    __bf16* Hb = Hws + (size_t)e * TOK * INTER;
#pragma unroll
    for (int i = 0; i < 2; ++i)
#pragma unroll
        for (int j = 0; j < 4; ++j)
#pragma unroll
            for (int r = 0; r < 8; ++r) {
                const int m = m0 + wm * 32 + i * 16 + half * 8 + r;
                const int n = n0 + wn * 64 + j * 16 + l16;
                const float g = accG[i][j][r];
                const float u = accU[i][j][r];
                const float s = g / (1.0f + __expf(-g));
                Hb[(size_t)m * INTER + n] = f2bf(u * s);
            }
}

// ---------------------------------------------------------------------------
// Kernel 2: out = h @ down_w.  grid = (H/BN, TOK/BM, E)
// A tile staged by the Tensor Data Mover: each wave DMAs a 16x32 bf16 slice.
// ---------------------------------------------------------------------------
__global__ __launch_bounds__(256)
void moe_down_kernel(const __bf16* __restrict__ Hws,
                     const float* __restrict__ Dw,
                     float* __restrict__ Out) {
    __shared__ __align__(32) __bf16 sA[2][BM * BK];   // row-major [m][k] (TDM)
    __shared__ __align__(32) __bf16 sB[2][BN * BK];   // transposed [n][k]

    const int e  = blockIdx.z;
    const int m0 = blockIdx.y * BM;
    const int n0 = blockIdx.x * BN;

    const __bf16* He = Hws + (size_t)e * TOK * INTER + (size_t)m0 * INTER;
    const float*  De = Dw  + (size_t)e * INTER * HID;

    const int t    = threadIdx.x;
    const int lane = t & 31;
    const int wave = t >> 5;
    const int wm   = wave & 3;
    const int wn   = wave >> 2;
    const int half = lane >> 4;
    const int l16  = lane & 15;

    const int bk4 = (t >> 5) << 2;
    const int bn4 = (t & 31) << 2;

    float dr[4][4];                   // B prefetch regs

    auto issue_tdmA = [&](int k0, int p) {     // per-wave 16-row DMA slice
        tdm_load_tile(He + (size_t)(wave * 16) * INTER + k0,
                      (unsigned)(uintptr_t)(&sA[p][wave * 16 * BK]),
                      16, INTER);
    };
    auto loadB = [&](int k0) {
#pragma unroll
        for (int r = 0; r < 4; ++r) {
            float4 d = *(const float4*)(De + (size_t)(k0 + bk4 + r) * HID + n0 + bn4);
            dr[r][0] = d.x; dr[r][1] = d.y; dr[r][2] = d.z; dr[r][3] = d.w;
        }
    };
    auto storeB = [&](int p) {
#pragma unroll
        for (int c = 0; c < 4; ++c) {
            v4bf d = { f2bf(dr[0][c]), f2bf(dr[1][c]), f2bf(dr[2][c]), f2bf(dr[3][c]) };
            *(v4bf*)(&sB[p][(bn4 + c) * BK + bk4]) = d;
        }
    };
    // A fragment from an unpermuted row: K{0..15} / K{16..31}
    auto loadAfrag = [&](int p, int m) -> v16bf {
        v8bf lo = *(const v8bf*)(&sA[p][m * BK + half * 16]);
        v8bf hi = *(const v8bf*)(&sA[p][m * BK + half * 16 + 8]);
        return __builtin_shufflevector(lo, hi, 0, 1, 2, 3, 4, 5, 6, 7,
                                       8, 9, 10, 11, 12, 13, 14, 15);
    };

    v8f acc[2][4];
#pragma unroll
    for (int i = 0; i < 2; ++i)
#pragma unroll
        for (int j = 0; j < 4; ++j) acc[i][j] = {};

    issue_tdmA(0, 0);
    loadB(0);
    int p = 0;
    for (int k0 = 0; k0 < INTER; k0 += BK) {
        storeB(p);
        __builtin_amdgcn_s_wait_tensorcnt(0);   // this wave's DMA slice landed
        __syncthreads();                        // all slices + B visible
        if (k0 + BK < INTER) {
            issue_tdmA(k0 + BK, p ^ 1);         // overlaps with WMMA below
            loadB(k0 + BK);
        }

        v16bf a0 = loadAfrag(p, wm * 32 + l16);
        v16bf a1 = loadAfrag(p, wm * 32 + 16 + l16);
#pragma unroll
        for (int j = 0; j < 4; ++j) {
            v16bf b = *(const v16bf*)(&sB[p][(wn * 64 + j * 16 + l16) * BK + half * 16]);
            acc[0][j] = wmma_bf16(a0, b, acc[0][j]);
            acc[1][j] = wmma_bf16(a1, b, acc[1][j]);
        }
        p ^= 1;
    }

    float* Ob = Out + (size_t)e * TOK * HID;
#pragma unroll
    for (int i = 0; i < 2; ++i)
#pragma unroll
        for (int j = 0; j < 4; ++j)
#pragma unroll
            for (int r = 0; r < 8; ++r) {
                const int m = m0 + wm * 32 + i * 16 + half * 8 + r;
                const int n = n0 + wn * 64 + j * 16 + l16;
                Ob[(size_t)m * HID + n] = acc[i][j][r];
            }
}

// ---------------------------------------------------------------------------
extern "C" void kernel_launch(void* const* d_in, const int* in_sizes, int n_in,
                              void* d_out, int out_size, void* d_ws, size_t ws_size,
                              hipStream_t stream) {
    const float* X  = (const float*)d_in[0];   // (E*TOK, H)
    const float* Gw = (const float*)d_in[1];   // (E, H, F)
    const float* Uw = (const float*)d_in[2];   // (E, H, F)
    const float* Dw = (const float*)d_in[3];   // (E, F, H)
    float* Out = (float*)d_out;                // (E*TOK, H)

    __bf16* Hws = (__bf16*)d_ws;                           // 128 MB h (bf16)
    __bf16* Xb  = Hws + (size_t)NUM_EXPERTS * TOK * INTER; // +32 MB X (bf16)

    const int xtot = NUM_EXPERTS * TOK * HID;
    cvt_x_kernel<<<xtot / (256 * 8), 256, 0, stream>>>(X, Xb);

    dim3 g1(INTER / BN, TOK / BM, NUM_EXPERTS);
    moe_gateup_kernel<<<g1, 256, 0, stream>>>(Xb, Gw, Uw, Hws);

    dim3 g2(HID / BN, TOK / BM, NUM_EXPERTS);
    moe_down_kernel<<<g2, 256, 0, stream>>>(Hws, Dw, Out);
}
